// DSVT_55387898249980
// MI455X (gfx1250) — matrix operand order repaired and checked by reading, not compiled, not run on hardware
//
#include <hip/hip_runtime.h>

// DSVT encoder shift (2 set partitions) for MI455X / gfx1250, wave32 + WMMA bf16.
// N=98280 voxels, D=192, DFF=384, H=8 heads of 24, L=36 per set, S=2730 sets.

#define N_VOX 98280
#define D 192
#define DFF 384
#define NH 8
#define HD 24
#define SL 36
#define S_SETS 2730
#define TD 576                      // 3*D
#define RB 32                       // rows per GEMM block (2 row-tiles share B frags)
#define ROWB 3072                   // ceil(N/32)

typedef __bf16 bf16;
typedef __attribute__((ext_vector_type(16))) __bf16 v16bf;
typedef __attribute__((ext_vector_type(8)))  float  v8f;

// ---- WMMA fragment helpers (contiguous k order k = 16*(lane/16) + s in A and B, wave32) ----------

// A matrix 16x32 bf16 tile from LDS (row-major, 'stride' elements per row).
// lanes 0-15: row=lane, K = 0..15; lanes 16-31: row=lane-16, K = 16..31
__device__ __forceinline__ v16bf frag_a_lds(const bf16* base, int stride, int k0, int lane) {
  int r  = lane & 15;
  int kh = (lane >> 4) << 4;        // 0 or 16
  v16bf f;
#pragma unroll
  for (int j = 0; j < 8; ++j) {
    int k = k0 + kh + (j << 1);
    f[2*j]     = base[r * stride + k];
    f[2*j + 1] = base[r * stride + k + 1];
  }
  return f;
}

// B matrix 32x16 tile, B = W^T with W row-major (out x in), already bf16.
// lanes 0-15: col=lane, K=k0+0..15; lanes 16-31: col=lane-16, K=k0+16..31
// Per lane this is 16 contiguous bf16 (32B) -> vectorizes to b128 loads.
__device__ __forceinline__ v16bf frag_b_bf(const bf16* W, int ld, int row0, int k0, int lane) {
  const bf16* p = W + (size_t)(row0 + (lane & 15)) * ld + k0 + ((lane >> 4) << 4);
  v16bf f;
#pragma unroll
  for (int j = 0; j < 16; ++j) f[j] = p[j];
  return f;
}

__device__ __forceinline__ v8f wmma_bf16(v16bf a, v16bf b, v8f c) {
  return __builtin_amdgcn_wmma_f32_16x16x32_bf16(false, a, false, b, (short)0, c, false, false);
}

// ---- Kernel 0: one-time f32 -> bf16 weight conversion ----------------------
__global__ __launch_bounds__(256)
void k_cvt(const float* __restrict__ src, bf16* __restrict__ dst, int n) {
  int i = blockIdx.x * 256 + threadIdx.x;
  if (i < n) dst[i] = (bf16)src[i];
}

// ---- Kernel 1: gathered QKV projection (bf16 WMMA, f32 accumulate) ---------
// Output: qkv[N][576] bf16, gathered (set-flattened) row order.
__global__ __launch_bounds__(384)
void k_qkv(const float* __restrict__ x, const float* __restrict__ pos,
           const int* __restrict__ inds,
           const bf16* __restrict__ Wqkv, const float* __restrict__ bqkv,
           bf16* __restrict__ qkv) {
  __shared__ bf16 qkL[RB * 194];    // x+pos (Q/K input), conflict-free stride
  __shared__ bf16 vL [RB * 194];    // x (V input)
  const int r0 = blockIdx.x * RB, tid = threadIdx.x;

  for (int i = tid; i < RB * D; i += 384) {
    int rr = i / D, c = i - rr * D;
    int g = r0 + rr;
    float xv = 0.f, pv = 0.f;
    if (g < N_VOX) { int vox = inds[g]; xv = x[(size_t)vox * D + c]; pv = pos[(size_t)vox * D + c]; }
    qkL[rr * 194 + c] = (bf16)(xv + pv);
    vL [rr * 194 + c] = (bf16)xv;
  }
  __syncthreads();

  const int wave = tid >> 5, lane = tid & 31;
#pragma unroll 1
  for (int t = 0; t < 3; ++t) {
    int ct = wave * 3 + t;                          // 36 col tiles / 12 waves
    const bf16* aL = (ct < 24) ? qkL : vL;          // Q,K tiles use x+pos; V tiles use x
    v8f acc0 = {0.f, 0.f, 0.f, 0.f, 0.f, 0.f, 0.f, 0.f};
    v8f acc1 = acc0;
#pragma unroll
    for (int kt = 0; kt < 6; ++kt) {
      v16bf b  = frag_b_bf(Wqkv, D, ct * 16, kt * 32, lane);   // shared by both row tiles
      v16bf a0 = frag_a_lds(aL,            194, kt * 32, lane);
      v16bf a1 = frag_a_lds(aL + 16 * 194, 194, kt * 32, lane);
      acc0 = wmma_bf16(a0, b, acc0);
      acc1 = wmma_bf16(a1, b, acc1);
    }
    int col = ct * 16 + (lane & 15);
    float bias = bqkv[col];
#pragma unroll
    for (int v = 0; v < 8; ++v) {
      int m = ((lane >> 4) << 3) + v;
      int g = r0 + m;
      if (g < N_VOX) qkv[(size_t)g * TD + col] = (bf16)(acc0[v] + bias);
      g = r0 + 16 + m;
      if (g < N_VOX) qkv[(size_t)g * TD + col] = (bf16)(acc1[v] + bias);
    }
  }
}

// ---- Kernel 2: per-set attention + fused O-projection + scatter ------------
__global__ __launch_bounds__(288)
void k_attn(const bf16* __restrict__ qkv, const int* __restrict__ inds,
            const unsigned char* __restrict__ mask,
            const bf16* __restrict__ Wo, const float* __restrict__ bo,
            float* __restrict__ src2) {
  __shared__ bf16 Qs[48 * 194];     // rows 36..47 zero; reused as attn-out for O-proj
  __shared__ bf16 Ks[36 * 194];
  __shared__ bf16 Vs[36 * 194];
  const int s = blockIdx.x, tid = threadIdx.x;
  const size_t base = (size_t)s * SL;

  for (int i = tid; i < SL * D; i += 288) {
    int r = i / D, c = i - r * D;
    size_t gr = (base + r) * TD;
    Qs[r * 194 + c] = qkv[gr + c];
    Ks[r * 194 + c] = qkv[gr + D + c];
    Vs[r * 194 + c] = qkv[gr + 2 * D + c];
  }
  for (int i = tid; i < 12 * 194; i += 288) Qs[36 * 194 + i] = (bf16)0.f;
  __syncthreads();

  const int l = tid / NH, h = tid - (tid / NH) * NH;   // 36x8 == 288 threads
  const float scale = 0.20412414523193154f;            // 1/sqrt(24)
  float sc[SL];
  float mx = -1e30f;
#pragma unroll
  for (int m = 0; m < SL; ++m) {
    float d = 0.f;
    const bf16* qp = &Qs[l * 194 + h * HD];
    const bf16* kp = &Ks[m * 194 + h * HD];
#pragma unroll
    for (int e = 0; e < HD; ++e) d += (float)qp[e] * (float)kp[e];
    d *= scale;
    if (mask[(size_t)s * SL + m]) d = -1e9f;
    sc[m] = d;
    mx = fmaxf(mx, d);
  }
  __syncthreads();                                     // everyone done reading Q/K

  float sum = 0.f;
#pragma unroll
  for (int m = 0; m < SL; ++m) { sc[m] = __expf(sc[m] - mx); sum += sc[m]; }
  float inv = 1.f / sum;
#pragma unroll 1
  for (int d = 0; d < HD; ++d) {
    float o = 0.f;
#pragma unroll
    for (int m = 0; m < SL; ++m) o += sc[m] * (float)Vs[m * 194 + h * HD + d];
    Qs[l * 194 + h * HD + d] = (bf16)(o * inv);        // attn-out into Q area
  }
  __syncthreads();

  // O-projection: (48x192)x(192x192), 3 row-tiles x 12 col-tiles over 9 waves
  const int wave = tid >> 5, lane = tid & 31;
#pragma unroll 1
  for (int j = 0; j < 4; ++j) {
    int job = wave * 4 + j;                            // 36 jobs
    int rt = job / 12, ct = job - rt * 12;
    v8f acc = {0.f, 0.f, 0.f, 0.f, 0.f, 0.f, 0.f, 0.f};
#pragma unroll
    for (int kt = 0; kt < 6; ++kt) {
      v16bf a = frag_a_lds(&Qs[rt * 16 * 194], 194, kt * 32, lane);
      v16bf b = frag_b_bf(Wo, D, ct * 16, kt * 32, lane);
      acc = wmma_bf16(a, b, acc);
    }
    int col = ct * 16 + (lane & 15);
    float bias = bo[col];
#pragma unroll
    for (int v = 0; v < 8; ++v) {
      int m = ((lane >> 4) << 3) + v;
      int row = rt * 16 + m;
      if (row < SL) {
        int vox = inds[base + row];                    // inds is a permutation: direct scatter
        src2[(size_t)vox * D + col] = acc[v] + bias;
      }
    }
  }
}

// ---- Kernel 3: h1 = LN(x + src2), wave-per-row -----------------------------
__global__ __launch_bounds__(256)
void k_addln(const float* __restrict__ x, const float* __restrict__ r,
             const float* __restrict__ g, const float* __restrict__ b,
             float* __restrict__ out) {
  const int row = blockIdx.x * 8 + (threadIdx.x >> 5);
  const int lane = threadIdx.x & 31;
  float v[6];
  float s = 0.f;
#pragma unroll
  for (int j = 0; j < 6; ++j) {
    int c = j * 32 + lane;
    v[j] = x[(size_t)row * D + c] + r[(size_t)row * D + c];
    s += v[j];
  }
#pragma unroll
  for (int o = 16; o > 0; o >>= 1) s += __shfl_xor(s, o, 32);
  float mean = s * (1.f / D);
  float q = 0.f;
#pragma unroll
  for (int j = 0; j < 6; ++j) { float d = v[j] - mean; q += d * d; }
#pragma unroll
  for (int o = 16; o > 0; o >>= 1) q += __shfl_xor(q, o, 32);
  float rs = rsqrtf(q * (1.f / D) + 1e-5f);
#pragma unroll
  for (int j = 0; j < 6; ++j) {
    int c = j * 32 + lane;
    out[(size_t)row * D + c] = (v[j] - mean) * rs * g[c] + b[c];
  }
}

// ---- Kernel 4: fused FFN: relu(h1@W1^T+b1)@W2^T+b2, 32 rows per block ------
__global__ __launch_bounds__(384)
void k_ffn(const float* __restrict__ h1,
           const bf16* __restrict__ W1, const float* __restrict__ b1,
           const bf16* __restrict__ W2, const float* __restrict__ b2,
           float* __restrict__ out) {
  __shared__ bf16 aL [RB * 194];
  __shared__ bf16 mid[RB * 386];
  const int r0 = blockIdx.x * RB, tid = threadIdx.x;
  for (int i = tid; i < RB * D; i += 384) {
    int rr = i / D, c = i - rr * D;
    int g = r0 + rr;
    aL[rr * 194 + c] = (bf16)((g < N_VOX) ? h1[(size_t)g * D + c] : 0.f);
  }
  __syncthreads();
  const int wave = tid >> 5, lane = tid & 31;
#pragma unroll 1
  for (int t = 0; t < 2; ++t) {
    int ct = wave + t * 12;                            // 24 mid col tiles
    v8f acc0 = {0.f, 0.f, 0.f, 0.f, 0.f, 0.f, 0.f, 0.f};
    v8f acc1 = acc0;
#pragma unroll
    for (int kt = 0; kt < 6; ++kt) {
      v16bf b  = frag_b_bf(W1, D, ct * 16, kt * 32, lane);
      v16bf a0 = frag_a_lds(aL,            194, kt * 32, lane);
      v16bf a1 = frag_a_lds(aL + 16 * 194, 194, kt * 32, lane);
      acc0 = wmma_bf16(a0, b, acc0);
      acc1 = wmma_bf16(a1, b, acc1);
    }
    int col = ct * 16 + (lane & 15);
    float bias = b1[col];
#pragma unroll
    for (int v = 0; v < 8; ++v) {
      int m = ((lane >> 4) << 3) + v;
      mid[m * 386 + col]        = (bf16)fmaxf(acc0[v] + bias, 0.f);
      mid[(m + 16) * 386 + col] = (bf16)fmaxf(acc1[v] + bias, 0.f);
    }
  }
  __syncthreads();
  {
    int ct = wave;                                     // 12 out col tiles
    v8f acc0 = {0.f, 0.f, 0.f, 0.f, 0.f, 0.f, 0.f, 0.f};
    v8f acc1 = acc0;
#pragma unroll
    for (int kt = 0; kt < 12; ++kt) {
      v16bf b  = frag_b_bf(W2, DFF, ct * 16, kt * 32, lane);
      v16bf a0 = frag_a_lds(mid,            386, kt * 32, lane);
      v16bf a1 = frag_a_lds(mid + 16 * 386, 386, kt * 32, lane);
      acc0 = wmma_bf16(a0, b, acc0);
      acc1 = wmma_bf16(a1, b, acc1);
    }
    int col = ct * 16 + (lane & 15);
    float bias = b2[col];
#pragma unroll
    for (int v = 0; v < 8; ++v) {
      int m = ((lane >> 4) << 3) + v;
      int g = r0 + m;
      if (g < N_VOX) out[(size_t)g * D + col] = acc0[v] + bias;
      g = r0 + 16 + m;
      if (g < N_VOX) out[(size_t)g * D + col] = acc1[v] + bias;
    }
  }
}

// ---- Kernel 5: out = LN(LN(h1+ff; g2,be2) + identity; gn,bn) ---------------
__global__ __launch_bounds__(256)
void k_dualln(const float* __restrict__ h1, const float* __restrict__ ff,
              const float* __restrict__ idn,
              const float* __restrict__ g2, const float* __restrict__ b2,
              const float* __restrict__ gn, const float* __restrict__ bn,
              float* __restrict__ out) {
  const int row = blockIdx.x * 8 + (threadIdx.x >> 5);
  const int lane = threadIdx.x & 31;
  float v[6];
  float s = 0.f;
#pragma unroll
  for (int j = 0; j < 6; ++j) {
    int c = j * 32 + lane;
    v[j] = h1[(size_t)row * D + c] + ff[(size_t)row * D + c];
    s += v[j];
  }
#pragma unroll
  for (int o = 16; o > 0; o >>= 1) s += __shfl_xor(s, o, 32);
  float mean = s * (1.f / D);
  float q = 0.f;
#pragma unroll
  for (int j = 0; j < 6; ++j) { float d = v[j] - mean; q += d * d; }
#pragma unroll
  for (int o = 16; o > 0; o >>= 1) q += __shfl_xor(q, o, 32);
  float rs = rsqrtf(q * (1.f / D) + 1e-5f);
  s = 0.f;
#pragma unroll
  for (int j = 0; j < 6; ++j) {
    int c = j * 32 + lane;
    v[j] = (v[j] - mean) * rs * g2[c] + b2[c] + idn[(size_t)row * D + c];
    s += v[j];
  }
#pragma unroll
  for (int o = 16; o > 0; o >>= 1) s += __shfl_xor(s, o, 32);
  mean = s * (1.f / D);
  q = 0.f;
#pragma unroll
  for (int j = 0; j < 6; ++j) { float d = v[j] - mean; q += d * d; }
#pragma unroll
  for (int o = 16; o > 0; o >>= 1) q += __shfl_xor(q, o, 32);
  rs = rsqrtf(q * (1.f / D) + 1e-5f);
#pragma unroll
  for (int j = 0; j < 6; ++j) {
    int c = j * 32 + lane;
    out[(size_t)row * D + c] = (v[j] - mean) * rs * gn[c] + bn[c];
  }
}

// ---------------------------------------------------------------------------
extern "C" void kernel_launch(void* const* d_in, const int* in_sizes, int n_in,
                              void* d_out, int out_size, void* d_ws, size_t ws_size,
                              hipStream_t stream) {
  const float* src      = (const float*)d_in[0];
  const int*   inds[2]  = {(const int*)d_in[1], (const int*)d_in[2]};
  const unsigned char* masks[2] = {(const unsigned char*)d_in[3], (const unsigned char*)d_in[4]};
  const float* pos[2]   = {(const float*)d_in[5], (const float*)d_in[6]};
  const float* Wqkv     = (const float*)d_in[7];
  const float* bqkv     = (const float*)d_in[8];
  const float* Wo       = (const float*)d_in[9];
  const float* bo       = (const float*)d_in[10];
  const float* W1       = (const float*)d_in[11];
  const float* b1       = (const float*)d_in[12];
  const float* W2       = (const float*)d_in[13];
  const float* b2       = (const float*)d_in[14];
  const float* g1       = (const float*)d_in[15];
  const float* be1      = (const float*)d_in[16];
  const float* g2       = (const float*)d_in[17];
  const float* be2      = (const float*)d_in[18];
  const float* gn       = (const float*)d_in[19];
  const float* bn       = (const float*)d_in[20];

  const int NQ = 2 * TD * D;        // Wqkv elements
  const int NO = 2 * D * D;         // Wo
  const int N1 = 2 * DFF * D;       // W1
  const int N2 = 2 * D * DFF;       // W2

  char* ws = (char*)d_ws;
  size_t off = 0;
  bf16* wq_bf = reinterpret_cast<bf16*>(ws + off); off += ((size_t)NQ * 2 + 255) & ~(size_t)255;
  bf16* wo_bf = reinterpret_cast<bf16*>(ws + off); off += ((size_t)NO * 2 + 255) & ~(size_t)255;
  bf16* w1_bf = reinterpret_cast<bf16*>(ws + off); off += ((size_t)N1 * 2 + 255) & ~(size_t)255;
  bf16* w2_bf = reinterpret_cast<bf16*>(ws + off); off += ((size_t)N2 * 2 + 255) & ~(size_t)255;
  bf16*  qkvbuf = reinterpret_cast<bf16*>(ws + off);
  off += ((size_t)N_VOX * TD * 2 + 255) & ~(size_t)255;      // 113 MB
  float* s2buf  = reinterpret_cast<float*>(ws + off);
  off += ((size_t)N_VOX * D * 4 + 255) & ~(size_t)255;       // 75.5 MB (src2 / ff)
  float* h1buf  = reinterpret_cast<float*>(ws + off);        // 75.5 MB

  // One-time (per launch) weight conversion to bf16: ~1.2 MB total.
  k_cvt<<<(NQ + 255) / 256, 256, 0, stream>>>(Wqkv, wq_bf, NQ);
  k_cvt<<<(NO + 255) / 256, 256, 0, stream>>>(Wo,   wo_bf, NO);
  k_cvt<<<(N1 + 255) / 256, 256, 0, stream>>>(W1,   w1_bf, N1);
  k_cvt<<<(N2 + 255) / 256, 256, 0, stream>>>(W2,   w2_bf, N2);

  float* out = (float*)d_out;
  const float* xin = src;
  for (int li = 0; li < 2; ++li) {
    const bf16*  Wq  = wq_bf + (size_t)li * TD * D;
    const float* bq  = bqkv  + (size_t)li * TD;
    const bf16*  Wol = wo_bf + (size_t)li * D * D;
    const float* bol = bo    + (size_t)li * D;
    const bf16*  W1l = w1_bf + (size_t)li * DFF * D;
    const float* b1l = b1    + (size_t)li * DFF;
    const bf16*  W2l = w2_bf + (size_t)li * D * DFF;
    const float* b2l = b2    + (size_t)li * D;

    k_qkv   <<<ROWB,      384, 0, stream>>>(xin, pos[li], inds[li], Wq, bq, qkvbuf);
    k_attn  <<<S_SETS,    288, 0, stream>>>(qkvbuf, inds[li], masks[li], Wol, bol, s2buf);
    k_addln <<<N_VOX / 8, 256, 0, stream>>>(xin, s2buf, g1 + (size_t)li * D, be1 + (size_t)li * D, h1buf);
    k_ffn   <<<ROWB,      384, 0, stream>>>(h1buf, W1l, b1l, W2l, b2l, s2buf);
    k_dualln<<<N_VOX / 8, 256, 0, stream>>>(h1buf, s2buf, xin,
                                            g2 + (size_t)li * D, be2 + (size_t)li * D,
                                            gn + (size_t)li * D, bn + (size_t)li * D, out);
    xin = out;
  }
  (void)in_sizes; (void)n_in; (void)out_size; (void)ws_size;
}
